// DeepFM_25563645346366
// MI455X (gfx1250) — matrix operand order repaired and checked by reading, not compiled, not run on hardware
//
#include <hip/hip_runtime.h>
#include <hip/hip_bf16.h>
#include <math.h>

// ---------------------------------------------------------------------------
// DeepFM forward on MI455X (gfx1250, wave32).
//   - gather + first/second-order FM: one wave per sample (bandwidth bound)
//   - MLP GEMMs via v_wmma_f32_16x16x32_f16 (f16 A/B, f32 accumulate)
//   - wave tile 32x80: each B fragment feeds 2 WMMAs (A-fragment reuse),
//     14 b128 loads per 10 WMMAs; compile-time strides so the K-walk folds
//     into 24-bit instruction offsets.
// ---------------------------------------------------------------------------

typedef __attribute__((ext_vector_type(8)))  _Float16 v8h;
typedef __attribute__((ext_vector_type(16))) _Float16 v16h;
typedef __attribute__((ext_vector_type(8)))  float    v8f;

#define B_SZ  16384
#define NF    50
#define KD    16
#define D_IN  800      // NF*KD, K of GEMM1, multiple of 32
#define H_DIM 400
#define H_PAD 416      // H_DIM padded to multiple of 32 for GEMM2 K-dim

#define WMMA_F16(A, Bf, Cacc) \
  __builtin_amdgcn_wmma_f32_16x16x32_f16(false, (A), false, (Bf), (short)0, (Cacc), false, false)

// --------------------------- weight prep -----------------------------------
// W0t[n][k] = (f16) W0[k][n]   (400 x 800)
__global__ __launch_bounds__(256) void prep_w0t_kernel(
    const float* __restrict__ W0, _Float16* __restrict__ W0t) {
  int i = blockIdx.x * 256 + threadIdx.x;
  if (i >= H_DIM * D_IN) return;
  int n = i / D_IN, k = i % D_IN;
  W0t[(size_t)n * D_IN + k] = (_Float16)W0[(size_t)k * H_DIM + n];
}

// W1t[n][k] = (f16) W1[k][n], zero-padded rows k=400..415   (400 x 416)
__global__ __launch_bounds__(256) void prep_w1t_kernel(
    const float* __restrict__ W1, _Float16* __restrict__ W1t) {
  int i = blockIdx.x * 256 + threadIdx.x;
  if (i >= H_DIM * H_PAD) return;
  int n = i / H_PAD, k = i % H_PAD;
  W1t[(size_t)n * H_PAD + k] =
      (k < H_DIM) ? (_Float16)W1[(size_t)k * H_DIM + n] : (_Float16)0.f;
}

// zero the pad columns 400..415 of H1 so GEMM2's K=416 loop is exact
__global__ __launch_bounds__(256) void pad_h1_kernel(_Float16* __restrict__ H1) {
  int i = blockIdx.x * 256 + threadIdx.x;
  if (i >= B_SZ * 16) return;
  int row = i >> 4, col = H_DIM + (i & 15);
  H1[(size_t)row * H_PAD + col] = (_Float16)0.f;
}

// ---------------------- gather + FM (first/second order) -------------------
// One wave per sample: 50 contiguous features, embedding rows are 64B.
// Writes X[b][800] (f16) for the MLP and fs[b] = first + bias + second.
__global__ __launch_bounds__(256) void gather_fm_kernel(
    const int*   __restrict__ feats,   const float* __restrict__ values,
    const float* __restrict__ weights, const float* __restrict__ embedding,
    const float* __restrict__ bias,
    _Float16* __restrict__ X, float* __restrict__ fs) {
  int lane = threadIdx.x & 31;
  int wave = threadIdx.x >> 5;
  int b = blockIdx.x * 8 + wave;

  float first = 0.f;
  float s1[KD], s2[KD];
#pragma unroll
  for (int k = 0; k < KD; ++k) { s1[k] = 0.f; s2[k] = 0.f; }

  for (int j = lane; j < NF; j += 32) {
    int   f = feats[b * NF + j];
    float v = values[b * NF + j];
    first += weights[f] * v;

    const float4* er = (const float4*)(embedding + (size_t)f * KD);
    float4 e0 = er[0], e1 = er[1], e2 = er[2], e3 = er[3];
    float e[KD] = {e0.x, e0.y, e0.z, e0.w, e1.x, e1.y, e1.z, e1.w,
                   e2.x, e2.y, e2.z, e2.w, e3.x, e3.y, e3.z, e3.w};

    v8h h0, h1;
#pragma unroll
    for (int k = 0; k < 8; ++k) {
      h0[k] = (_Float16)e[k];
      h1[k] = (_Float16)e[k + 8];
    }
    v8h* xp = (v8h*)(X + (size_t)b * D_IN + j * KD);
    xp[0] = h0; xp[1] = h1;

#pragma unroll
    for (int k = 0; k < KD; ++k) {
      float ev = e[k] * v;
      s1[k] += ev;
      s2[k] += ev * ev;
    }
  }

#pragma unroll
  for (int off = 16; off > 0; off >>= 1) {
    first += __shfl_down(first, off, 32);
#pragma unroll
    for (int k = 0; k < KD; ++k) {
      s1[k] += __shfl_down(s1[k], off, 32);
      s2[k] += __shfl_down(s2[k], off, 32);
    }
  }

  if (lane == 0) {
    float second = 0.f;
#pragma unroll
    for (int k = 0; k < KD; ++k) second += s1[k] * s1[k] - s2[k];
    fs[b] = first + bias[0] + 0.5f * second;
  }
}

// ------------------------------ WMMA GEMM ----------------------------------
// C[M x N] = relu(A[M x K] * B[K x N] + bias), f16 in / f32 acc / f16 out.
// Bt is B transposed ([N][LDB]) so each lane's B fragment is one 32B load.
// Wave tile 32x80: 2 A fragments x 5 B fragments -> 10 accumulators.
// Block = 8 waves = 256 rows x 80 cols; grid (M/256, N/80).
// Fragment layouts used here (A/B share this code's contiguous K order):
//   A (16x32 f16): lane r=lane&15 holds row r; K run = half*16 .. +15 (contig)
//   B (32x16 f16): lane holds col lane&15; K run = half*16 .. +15 (contig)
//   C/D f32: VGPR j -> row j + 8*half, col = lane&15
template <int K, int LDA, int LDB, int LDO>
__global__ __launch_bounds__(256) void gemm_relu_f16_kernel(
    const _Float16* __restrict__ A, const _Float16* __restrict__ Bt,
    _Float16* __restrict__ C, const float* __restrict__ biasPtr) {
  int lane = threadIdx.x & 31;
  int wave = threadIdx.x >> 5;
  int r    = lane & 15;
  int half = lane >> 4;
  int row0 = blockIdx.x * 256 + wave * 32;
  int colb = blockIdx.y * 80;

  const _Float16* arow0 = A + (size_t)(row0 + r) * LDA + half * 16;
  const _Float16* arow1 = arow0 + (size_t)16 * LDA;
  const _Float16* bc0   = Bt + (size_t)(colb +  0 + r) * LDB + half * 16;
  const _Float16* bc1   = Bt + (size_t)(colb + 16 + r) * LDB + half * 16;
  const _Float16* bc2   = Bt + (size_t)(colb + 32 + r) * LDB + half * 16;
  const _Float16* bc3   = Bt + (size_t)(colb + 48 + r) * LDB + half * 16;
  const _Float16* bc4   = Bt + (size_t)(colb + 64 + r) * LDB + half * 16;

  v8f c00 = {}, c01 = {}, c02 = {}, c03 = {}, c04 = {};
  v8f c10 = {}, c11 = {}, c12 = {}, c13 = {}, c14 = {};

  for (int kk = 0; kk < K; kk += 32) {
    union { v16h v; v8h h[2]; } a0u, a1u;
    a0u.h[0] = *(const v8h*)(arow0 + kk);        // K = kk + half*16 .. +7
    a0u.h[1] = *(const v8h*)(arow0 + kk + 8);    // K = kk + half*16 + 8 ..
    a1u.h[0] = *(const v8h*)(arow1 + kk);
    a1u.h[1] = *(const v8h*)(arow1 + kk + 8);

    v16h b0 = *(const v16h*)(bc0 + kk);          // 32B contiguous in K
    v16h b1 = *(const v16h*)(bc1 + kk);
    v16h b2 = *(const v16h*)(bc2 + kk);
    v16h b3 = *(const v16h*)(bc3 + kk);
    v16h b4 = *(const v16h*)(bc4 + kk);

    c00 = WMMA_F16(a0u.v, b0, c00);
    c10 = WMMA_F16(a1u.v, b0, c10);
    c01 = WMMA_F16(a0u.v, b1, c01);
    c11 = WMMA_F16(a1u.v, b1, c11);
    c02 = WMMA_F16(a0u.v, b2, c02);
    c12 = WMMA_F16(a1u.v, b2, c12);
    c03 = WMMA_F16(a0u.v, b3, c03);
    c13 = WMMA_F16(a1u.v, b3, c13);
    c04 = WMMA_F16(a0u.v, b4, c04);
    c14 = WMMA_F16(a1u.v, b4, c14);
  }

  float bias = biasPtr[0];
  int rb0 = row0 + half * 8;        // rows rb0+j   (j=0..7) for c0x
  int rb1 = rb0 + 16;               // rows rb1+j   for c1x
#pragma unroll
  for (int j = 0; j < 8; ++j) {
    size_t o0 = (size_t)(rb0 + j) * LDO + colb + r;
    size_t o1 = (size_t)(rb1 + j) * LDO + colb + r;
    float v;
    v = c00[j] + bias; C[o0 +  0] = (_Float16)(v > 0.f ? v : 0.f);
    v = c01[j] + bias; C[o0 + 16] = (_Float16)(v > 0.f ? v : 0.f);
    v = c02[j] + bias; C[o0 + 32] = (_Float16)(v > 0.f ? v : 0.f);
    v = c03[j] + bias; C[o0 + 48] = (_Float16)(v > 0.f ? v : 0.f);
    v = c04[j] + bias; C[o0 + 64] = (_Float16)(v > 0.f ? v : 0.f);
    v = c10[j] + bias; C[o1 +  0] = (_Float16)(v > 0.f ? v : 0.f);
    v = c11[j] + bias; C[o1 + 16] = (_Float16)(v > 0.f ? v : 0.f);
    v = c12[j] + bias; C[o1 + 32] = (_Float16)(v > 0.f ? v : 0.f);
    v = c13[j] + bias; C[o1 + 48] = (_Float16)(v > 0.f ? v : 0.f);
    v = c14[j] + bias; C[o1 + 64] = (_Float16)(v > 0.f ? v : 0.f);
  }
}

// -------------------------- final matvec + sigmoid -------------------------
__global__ __launch_bounds__(256) void final_kernel(
    const _Float16* __restrict__ H2, const float* __restrict__ W2,
    const float* __restrict__ b2, const float* __restrict__ fs,
    float* __restrict__ out) {
  int lane = threadIdx.x & 31;
  int wave = threadIdx.x >> 5;
  int b = blockIdx.x * 8 + wave;

  const _Float16* h = H2 + (size_t)b * H_DIM;
  float dot = 0.f;
  for (int k = lane; k < H_DIM; k += 32) dot += (float)h[k] * W2[k];
#pragma unroll
  for (int off = 16; off > 0; off >>= 1) dot += __shfl_down(dot, off, 32);

  if (lane == 0) {
    float hi = dot + b2[0];
    hi = hi > 0.f ? hi : 0.f;
    float z = fs[b] + hi;
    out[b] = 1.f / (1.f + expf(-z));
  }
}

// ------------------------------- launcher ----------------------------------
extern "C" void kernel_launch(void* const* d_in, const int* in_sizes, int n_in,
                              void* d_out, int out_size, void* d_ws, size_t ws_size,
                              hipStream_t stream) {
  (void)in_sizes; (void)n_in; (void)out_size; (void)ws_size;
  // setup_inputs order: index, feats, values, bias, weights, embedding,
  //                     W0, b0, W1, b1, W2, b2, batch_size
  const int*   feats     = (const int*)  d_in[1];
  const float* values    = (const float*)d_in[2];
  const float* bias      = (const float*)d_in[3];
  const float* weights   = (const float*)d_in[4];
  const float* embedding = (const float*)d_in[5];
  const float* W0        = (const float*)d_in[6];
  const float* b0        = (const float*)d_in[7];
  const float* W1        = (const float*)d_in[8];
  const float* b1        = (const float*)d_in[9];
  const float* W2        = (const float*)d_in[10];
  const float* b2        = (const float*)d_in[11];
  float* out = (float*)d_out;

  char* ws = (char*)d_ws;
  size_t off = 0;
  auto take = [&](size_t bytes) -> char* {
    char* p = ws + off;
    off += (bytes + 255) & ~(size_t)255;   // keep 256B alignment for v16h loads
    return p;
  };
  _Float16* X   = (_Float16*)take((size_t)B_SZ  * D_IN  * 2);  // 26.2 MB
  _Float16* W0t = (_Float16*)take((size_t)H_DIM * D_IN  * 2);  // 0.64 MB
  _Float16* W1t = (_Float16*)take((size_t)H_DIM * H_PAD * 2);  // 0.33 MB
  _Float16* H1  = (_Float16*)take((size_t)B_SZ  * H_PAD * 2);  // 13.6 MB
  _Float16* H2  = (_Float16*)take((size_t)B_SZ  * H_DIM * 2);  // 13.1 MB
  float*    fs  = (float*)   take((size_t)B_SZ  * 4);          // 64 KB

  prep_w0t_kernel<<<(H_DIM * D_IN  + 255) / 256, 256, 0, stream>>>(W0, W0t);
  prep_w1t_kernel<<<(H_DIM * H_PAD + 255) / 256, 256, 0, stream>>>(W1, W1t);
  pad_h1_kernel  <<<(B_SZ * 16     + 255) / 256, 256, 0, stream>>>(H1);

  gather_fm_kernel<<<B_SZ / 8, 256, 0, stream>>>(feats, values, weights,
                                                 embedding, bias, X, fs);

  dim3 g(B_SZ / 256, H_DIM / 80);  // 64 x 5 workgroups, 8 waves each
  gemm_relu_f16_kernel<D_IN,  D_IN,  D_IN,  H_PAD><<<g, 256, 0, stream>>>(X,  W0t, H1, b0);
  gemm_relu_f16_kernel<H_PAD, H_PAD, H_PAD, H_DIM><<<g, 256, 0, stream>>>(H1, W1t, H2, b1);

  final_kernel<<<B_SZ / 8, 256, 0, stream>>>(H2, W2, b2, fs, out);
}
